// ConvLinear4bit_83476984365423
// MI455X (gfx1250) — 8-bit matrix operand order repaired and checked by reading, compiled, not run on hardware
//
#include <hip/hip_runtime.h>

typedef int v8i __attribute__((ext_vector_type(8)));

#define D          4096
#define ROTSZ      256
#define KSLAB      128
#define KSTEPS     (D / KSLAB)
#define LDS_STRIDE 144

__device__ __forceinline__ void async_b128(unsigned lds_addr, const void* gaddr) {
  asm volatile("global_load_async_to_lds_b128 %0, %1, off"
               :: "v"(lds_addr), "v"(gaddr)
               : "memory");
}
__device__ __forceinline__ void wait_asynccnt0() {
  asm volatile("s_wait_asynccnt 0x0" ::: "memory");
}
__device__ __forceinline__ unsigned lds_off(const void* p) {
  return (unsigned)(unsigned long long)p;
}

__global__ __launch_bounds__(256) void fwht_quant_kernel(
    const float* __restrict__ x, char* __restrict__ q8, float* __restrict__ sx)
{
  __shared__ float row[D];
  __shared__ float red[256];
  const int r = blockIdx.x;
  const int t = threadIdx.x;

  const float4* xr = (const float4*)(x + (size_t)r * D);
  float4* rr = (float4*)row;
#pragma unroll
  for (int j = 0; j < 4; ++j) rr[t + j * 256] = xr[t + j * 256];
  __syncthreads();

  for (int s = 1; s < ROTSZ; s <<= 1) {
#pragma unroll
    for (int j = 0; j < 8; ++j) {
      int p = t + j * 256;
      int i = ((p & ~(s - 1)) << 1) | (p & (s - 1));
      float a = row[i], b = row[i + s];
      row[i]     = a + b;
      row[i + s] = a - b;
    }
    __syncthreads();
  }

  float m = 0.0f;
#pragma unroll
  for (int j = 0; j < 16; ++j) m = fmaxf(m, fabsf(row[t * 16 + j]));
  red[t] = m;
  __syncthreads();
  for (int w = 128; w > 0; w >>= 1) {
    if (t < w) red[t] = fmaxf(red[t], red[t + w]);
    __syncthreads();
  }
  const float amax = red[0];
  const float inv  = amax > 0.0f ? 7.0f / amax : 0.0f;
  if (t == 0) sx[r] = amax * (1.0f / 112.0f);

  int* qo = (int*)(q8 + (size_t)r * D);
#pragma unroll
  for (int j = 0; j < 4; ++j) {
    int packed = 0;
#pragma unroll
    for (int b = 0; b < 4; ++b) {
      float v = row[t * 16 + j * 4 + b];
      int qi = (int)rintf(v * inv);
      qi = qi < -8 ? -8 : (qi > 7 ? 7 : qi);
      packed |= (qi & 0xFF) << (8 * b);
    }
    qo[t * 4 + j] = packed;
  }
}

__global__ __launch_bounds__(256) void pack_w_kernel(
    const int* __restrict__ w, char* __restrict__ w8)
{
  const size_t idx = (size_t)blockIdx.x * 256 + threadIdx.x;
  const int4 v = ((const int4*)w)[idx];
  int packed = (v.x & 0xFF) | ((v.y & 0xFF) << 8) |
               ((v.z & 0xFF) << 16) | ((v.w & 0xFF) << 24);
  ((int*)w8)[idx] = packed;
}

__global__ __launch_bounds__(256) void gemm_iu8_kernel(
    const char* __restrict__ qx, const char* __restrict__ w8,
    const float* __restrict__ sx, const float* __restrict__ wsc,
    const float* __restrict__ bias, float* __restrict__ out)
{
  __shared__ char ldsA[2][128 * LDS_STRIDE];
  __shared__ char ldsB[2][128 * LDS_STRIDE];

  const int t     = threadIdx.x;
  const int lane  = t & 31;
  const int wave  = t >> 5;
  const int waveM = wave >> 1;
  const int waveN = wave & 1;
  const int m     = lane & 15;
  const int khalf = lane >> 4;

  const int tileM = blockIdx.y * 128;
  const int tileN = blockIdx.x * 128;

  const int lrow = t >> 1;
  const int loff = (t & 1) * 64;

  const char* gA = qx + (size_t)(tileM + lrow) * D + loff;
  const char* gB = w8 + (size_t)(tileN + lrow) * D + loff;

  const unsigned aL[2] = { lds_off(&ldsA[0][0]) + (unsigned)(lrow * LDS_STRIDE + loff),
                           lds_off(&ldsA[1][0]) + (unsigned)(lrow * LDS_STRIDE + loff) };
  const unsigned bL[2] = { lds_off(&ldsB[0][0]) + (unsigned)(lrow * LDS_STRIDE + loff),
                           lds_off(&ldsB[1][0]) + (unsigned)(lrow * LDS_STRIDE + loff) };

#pragma unroll
  for (int q = 0; q < 4; ++q) {
    async_b128(aL[0] + q * 16, gA + q * 16);
    async_b128(bL[0] + q * 16, gB + q * 16);
  }

  v8i acc[2][4] = {};

#pragma unroll 1
  for (int kk = 0; kk < KSTEPS; ++kk) {
    wait_asynccnt0();
    __syncthreads();

    if (kk + 1 < KSTEPS) {
      const int nb = (kk + 1) & 1;
      const char* ga = gA + (kk + 1) * KSLAB;
      const char* gb = gB + (kk + 1) * KSLAB;
#pragma unroll
      for (int q = 0; q < 4; ++q) {
        async_b128(aL[nb] + q * 16, ga + q * 16);
        async_b128(bL[nb] + q * 16, gb + q * 16);
      }
    }

    const char* sA = ldsA[kk & 1];
    const char* sB = ldsB[kk & 1];

#pragma unroll
    for (int ks = 0; ks < 2; ++ks) {
      const int kb = ks * 64;
      v8i a[2], b[4];
#pragma unroll
      for (int tm = 0; tm < 2; ++tm) {
        const char* base = sA + (waveM * 32 + tm * 16 + m) * LDS_STRIDE + kb;
#pragma unroll
        for (int g = 0; g < 4; ++g) {
          int2 d = *(const int2*)(base + (g >> 1) * 32 + khalf * 16 + (g & 1) * 8);
          a[tm][2 * g]     = d.x;
          a[tm][2 * g + 1] = d.y;
        }
      }
#pragma unroll
      for (int tn = 0; tn < 4; ++tn) {
        const char* base = sB + (waveN * 64 + tn * 16 + m) * LDS_STRIDE + kb;
        int4 lo = *(const int4*)(base + khalf * 16);
        int4 hi = *(const int4*)(base + 32 + khalf * 16);
        b[tn][0] = lo.x; b[tn][1] = lo.y; b[tn][2] = lo.z; b[tn][3] = lo.w;
        b[tn][4] = hi.x; b[tn][5] = hi.y; b[tn][6] = hi.z; b[tn][7] = hi.w;
      }

#pragma unroll
      for (int tm = 0; tm < 2; ++tm)
#pragma unroll
        for (int tn = 0; tn < 4; ++tn)
          acc[tm][tn] = __builtin_amdgcn_wmma_i32_16x16x64_iu8(
              true, a[tm], true, b[tn], acc[tm][tn], false, false);
    }
  }

#pragma unroll
  for (int tm = 0; tm < 2; ++tm) {
    const int rbase = tileM + waveM * 32 + tm * 16 + khalf * 8;
#pragma unroll
    for (int tn = 0; tn < 4; ++tn) {
      const int gcol = tileN + waveN * 64 + tn * 16 + m;
      const float wc = wsc[gcol];
      const float bi = bias[gcol];
#pragma unroll
      for (int rv = 0; rv < 8; ++rv) {
        const int grow = rbase + rv;
        out[(size_t)grow * D + gcol] =
            (float)acc[tm][tn][rv] * sx[grow] * wc + bi;
      }
    }
  }
}

extern "C" void kernel_launch(void* const* d_in, const int* in_sizes, int n_in,
                              void* d_out, int out_size, void* d_ws, size_t ws_size,
                              hipStream_t stream) {
  const float* x    = (const float*)d_in[0];
  const int*   wint = (const int*)d_in[1];
  const float* wsc  = (const float*)d_in[2];
  const float* bias = (const float*)d_in[3];
  float* out = (float*)d_out;

  char*  q8 = (char*)d_ws;
  char*  w8 = q8 + (size_t)D * D;
  float* sx = (float*)(w8 + (size_t)D * D);

  fwht_quant_kernel<<<D, 256, 0, stream>>>(x, q8, sx);
  pack_w_kernel<<<(D * (D / 4)) / 256, 256, 0, stream>>>(wint, w8);

  dim3 grid(D / 128, D / 128);
  gemm_iu8_kernel<<<grid, 256, 0, stream>>>(q8, w8, sx, wsc, bias, out);
}
